// VQVAE_43937515438428
// MI455X (gfx1250) — compile-verified, matrix operand order checked by reading, not run on hardware
//
#include <hip/hip_runtime.h>
#include <hip/hip_bf16.h>

// ---------------------------------------------------------------------------
// VQ-VAE forward for MI455X (gfx1250): implicit-GEMM convs on bf16 WMMA,
// double-buffered LDS pipeline, async-to-LDS weight staging when available.
// ---------------------------------------------------------------------------

typedef unsigned short u16b;
typedef __attribute__((ext_vector_type(16))) __bf16 v16bf;
typedef __attribute__((ext_vector_type(8)))  float  v8f;
typedef int v4i __attribute__((vector_size(16)));   // matches builtin param type

#define AS1 __attribute__((address_space(1)))
#define AS3 __attribute__((address_space(3)))

#if __has_builtin(__builtin_amdgcn_global_load_async_to_lds_b128)
#define ASYNC_LDS 1
#else
#define ASYNC_LDS 0
#endif

union FragU {
    uint4 q[2];   // 32 bytes
    v16bf v;      // 16 bf16 = 32 bytes
};

__device__ __forceinline__ float bf2f(u16b h) {
    unsigned u = ((unsigned)h) << 16;
    return __builtin_bit_cast(float, u);
}
__device__ __forceinline__ u16b f2bf(float f) {
    unsigned u = __builtin_bit_cast(unsigned, f);
    unsigned r = u + 0x7FFFu + ((u >> 16) & 1u);   // round-to-nearest-even
    return (u16b)(r >> 16);
}
// packed bf16x2 ReLU: zero any half whose sign bit is set
__device__ __forceinline__ unsigned relu_pk(unsigned w) {
    unsigned s = w & 0x80008000u;
    return w & ~((s >> 15) * 0xFFFFu);
}
__device__ __forceinline__ void wait_async0() {
#if __has_builtin(__builtin_amdgcn_s_wait_asynccnt)
    __builtin_amdgcn_s_wait_asynccnt(0);
#else
    asm volatile("s_wait_asynccnt 0x0" ::: "memory");
#endif
}

// ---------------------------------------------------------------------------
// Implicit GEMM conv kernel.
//   A: activations NHWC bf16 (N, IH, IW, Cin)
//   W: weights bf16, row-major [Cout][Ktot], Ktot = KH*KW*Cin, k = tap*Cin+ci
//   TRANS=1: transposed conv, stride 2, pad 1 (output-gather formulation)
// Tile: BM=128 x BN=128 x BK=32; 256 threads = 8 wave32, each wave 32x64.
// ---------------------------------------------------------------------------
struct GP {
    const u16b* A;
    const u16b* W;
    const float* bias;
    const u16b* R;      // residual (NHWC bf16) or nullptr
    u16b* D;            // output NHWC bf16 (N, OH, OW, Cout)
    int N, IH, IW, Cin, OH, OW, Cout, KH, KW, stride, pad;
    int Ktot, Mtot, relu_in, relu_out;
};

template <int TRANS>
__global__ void __launch_bounds__(256) k_conv_gemm(GP p) {
    __shared__ __align__(16) u16b lA[2][128 * 40];   // +8 elem pad per row
    __shared__ __align__(16) u16b lB[2][128 * 40];

    const int tid = threadIdx.x;
    const int m0  = blockIdx.x * 128;
    const int n0  = blockIdx.y * 128;

    // Each thread stages one half-row (16 consecutive k = 32 bytes).
    const int mm = tid >> 1;
    const int kb = (tid & 1) << 4;

    const int  gm     = m0 + mm;
    const bool mvalid = (gm < p.Mtot);
    int an = 0, aoy = 0, aox = 0;
    if (mvalid) {
        const int hw = p.OH * p.OW;
        an = gm / hw;
        int rem = gm - an * hw;
        aoy = rem / p.OW;
        aox = rem - aoy * p.OW;
    }
    const int  gn     = n0 + mm;
    const bool nvalid = (gn < p.Cout);

    const bool fastA = (p.Cin & 31) == 0;            // tap uniform per k-step
    const bool fastB = fastA && ((p.Cout & 127) == 0);

    // Wave tiling: 4 waves along M, 2 along N.
    const int wid  = tid >> 5;
    const int lane = tid & 31;
    const int wm   = (wid & 3) << 5;    // 0,32,64,96
    const int wn   = (wid >> 2) << 6;   // 0,64
    const int lrow = lane & 15;
    const int klo  = (lane >> 4) << 3;  // K sub-offset 0 or 8 per 16-bit A layout

    v8f acc[2][4];
    const v8f vzero = {0.f, 0.f, 0.f, 0.f, 0.f, 0.f, 0.f, 0.f};
#pragma unroll
    for (int i = 0; i < 2; ++i)
#pragma unroll
        for (int j = 0; j < 4; ++j) acc[i][j] = vzero;

    // ---- staging helpers ----
    auto stageA = [&](int sb, int k0) {
        u16b* l = &lA[sb][mm * 40 + kb];
        if (fastA) {
            const int tap = k0 / p.Cin;          // uniform across block
            const int cb  = k0 - tap * p.Cin;
            const int ky  = tap / p.KW;
            const int kx  = tap - ky * p.KW;
            bool v = mvalid;
            size_t idx = 0;
            if (!TRANS) {
                const int iy = aoy * p.stride - p.pad + ky;
                const int ix = aox * p.stride - p.pad + kx;
                v = v && ((unsigned)iy < (unsigned)p.IH) && ((unsigned)ix < (unsigned)p.IW);
                idx = (((size_t)an * p.IH + iy) * p.IW + ix) * p.Cin + cb + kb;
            } else {
                const int ty = aoy + p.pad - ky;
                const int tx = aox + p.pad - kx;
                v = v && ((ty | tx) >= 0) && !((ty | tx) & 1);
                const int iy = ty >> 1, ix = tx >> 1;
                v = v && (iy < p.IH) && (ix < p.IW);
                idx = (((size_t)an * p.IH + iy) * p.IW + ix) * p.Cin + cb + kb;
            }
            uint4 q0 = {0, 0, 0, 0}, q1 = {0, 0, 0, 0};
            if (v) {
                const uint4* gp = (const uint4*)(p.A + idx);
                q0 = gp[0];
                q1 = gp[1];
            }
            if (p.relu_in) {
                q0.x = relu_pk(q0.x); q0.y = relu_pk(q0.y);
                q0.z = relu_pk(q0.z); q0.w = relu_pk(q0.w);
                q1.x = relu_pk(q1.x); q1.y = relu_pk(q1.y);
                q1.z = relu_pk(q1.z); q1.w = relu_pk(q1.w);
            }
            *(uint4*)l       = q0;
            *(uint4*)(l + 8) = q1;
        } else {
#pragma unroll
            for (int i = 0; i < 16; ++i) {
                const int k = k0 + kb + i;
                u16b v = 0;
                if (mvalid && k < p.Ktot) {
                    const int tap = k / p.Cin;
                    const int ci  = k - tap * p.Cin;
                    const int ky  = tap / p.KW;
                    const int kx  = tap - ky * p.KW;
                    if (!TRANS) {
                        const int iy = aoy * p.stride - p.pad + ky;
                        const int ix = aox * p.stride - p.pad + kx;
                        if ((unsigned)iy < (unsigned)p.IH && (unsigned)ix < (unsigned)p.IW)
                            v = p.A[(((size_t)an * p.IH + iy) * p.IW + ix) * p.Cin + ci];
                    } else {
                        const int ty = aoy + p.pad - ky;
                        const int tx = aox + p.pad - kx;
                        if (((ty | tx) >= 0) && !((ty | tx) & 1)) {
                            const int iy = ty >> 1, ix = tx >> 1;
                            if (iy < p.IH && ix < p.IW)
                                v = p.A[(((size_t)an * p.IH + iy) * p.IW + ix) * p.Cin + ci];
                        }
                    }
                    if (p.relu_in && (v & 0x8000u)) v = 0;
                }
                l[i] = v;
            }
        }
    };

    auto stageB = [&](int sb, int k0) {
        u16b* l = &lB[sb][mm * 40 + kb];
        if (fastB) {
            const u16b* g = p.W + (size_t)gn * p.Ktot + k0 + kb;
#if ASYNC_LDS
            // gfx1250 async DMA into LDS, tracked by ASYNCcnt
            __builtin_amdgcn_global_load_async_to_lds_b128(
                (AS1 v4i*)g, (AS3 v4i*)l, 0, 0);
            __builtin_amdgcn_global_load_async_to_lds_b128(
                (AS1 v4i*)(g + 8), (AS3 v4i*)(l + 8), 0, 0);
#else
            const uint4* gp = (const uint4*)g;
            *(uint4*)l       = gp[0];
            *(uint4*)(l + 8) = gp[1];
#endif
        } else {
#pragma unroll
            for (int i = 0; i < 16; ++i) {
                const int k = k0 + kb + i;
                u16b v = 0;
                if (nvalid && k < p.Ktot) v = p.W[(size_t)gn * p.Ktot + k];
                l[i] = v;
            }
            if (nvalid && (k0 + 32 < p.Ktot))
                __builtin_prefetch(p.W + (size_t)gn * p.Ktot + k0 + 32, 0, 1);
        }
    };

    auto compute = [&](int sb) {
        FragU a[2], b[4];
#pragma unroll
        for (int i = 0; i < 2; ++i) {
            const int r = wm + i * 16 + lrow;
            a[i].q[0] = *(const uint4*)&lA[sb][r * 40 + klo];
            a[i].q[1] = *(const uint4*)&lA[sb][r * 40 + klo + 16];
        }
#pragma unroll
        for (int j = 0; j < 4; ++j) {
            const int r = wn + j * 16 + lrow;
            b[j].q[0] = *(const uint4*)&lB[sb][r * 40 + klo];
            b[j].q[1] = *(const uint4*)&lB[sb][r * 40 + klo + 16];
        }
#pragma unroll
        for (int i = 0; i < 2; ++i)
#pragma unroll
            for (int j = 0; j < 4; ++j)
                acc[i][j] = __builtin_amdgcn_wmma_f32_16x16x32_bf16(
                    false, a[i].v, false, b[j].v, (short)0, acc[i][j], false, false);
    };

    // ---- software-pipelined main loop (double-buffered LDS) ----
    stageA(0, 0);
    stageB(0, 0);
    int buf = 0;
    for (int k0 = 0; k0 < p.Ktot; k0 += 32) {
        if (fastB) wait_async0();
        __syncthreads();
        if (k0 + 32 < p.Ktot) {
            stageA(buf ^ 1, k0 + 32);
            stageB(buf ^ 1, k0 + 32);
        }
        compute(buf);
        buf ^= 1;
    }

    // ---- epilogue: bias, residual, relu, store bf16 NHWC ----
    const int cN    = lane & 15;
    const int mhalf = (lane >> 4) << 3;   // C/D layout: lanes 16-31 hold M=r+8
#pragma unroll
    for (int i = 0; i < 2; ++i) {
#pragma unroll
        for (int j = 0; j < 4; ++j) {
#pragma unroll
            for (int r = 0; r < 8; ++r) {
                const int gmo = m0 + wm + i * 16 + mhalf + r;
                const int gno = n0 + wn + j * 16 + cN;
                if (gmo < p.Mtot && gno < p.Cout) {
                    float v = acc[i][j][r] + p.bias[gno];
                    if (p.R) v += bf2f(p.R[(size_t)gmo * p.Cout + gno]);
                    if (p.relu_out) v = fmaxf(v, 0.f);
                    p.D[(size_t)gmo * p.Cout + gno] = f2bf(v);
                }
            }
        }
    }
}

// ---------------------------------------------------------------------------
// Layout / weight conversion kernels
// ---------------------------------------------------------------------------
__global__ void k_tonhwc(const float* src, u16b* dst, int N, int C, int H, int W) {
    int i = blockIdx.x * blockDim.x + threadIdx.x;
    int total = N * H * W * C;
    if (i >= total) return;
    int c = i % C; int t = i / C;
    int w = t % W; t /= W;
    int h = t % H; int n = t / H;
    dst[i] = f2bf(src[(((size_t)n * C + c) * H + h) * W + w]);
}

__global__ void k_tonchw(const u16b* src, float* dst, int N, int C, int H, int W) {
    int i = blockIdx.x * blockDim.x + threadIdx.x;
    int total = N * C * H * W;
    if (i >= total) return;
    int w = i % W; int t = i / W;
    int h = t % H; t /= H;
    int c = t % C; int n = t / C;
    dst[i] = bf2f(src[(((size_t)n * H + h) * W + w) * C + c]);
}

// OIHW fp32 -> bf16 [Cout][tap*Cin+ci]
__global__ void k_wconv(const float* src, u16b* dst, int Cout, int Cin, int KH, int KW) {
    int i = blockIdx.x * blockDim.x + threadIdx.x;
    int K = KH * KW * Cin;
    if (i >= Cout * K) return;
    int n = i / K; int k = i - n * K;
    int tap = k / Cin; int ci = k - tap * Cin;
    int ky = tap / KW; int kx = tap - ky * KW;
    dst[i] = f2bf(src[(((size_t)n * Cin + ci) * KH + ky) * KW + kx]);
}

// ConvTranspose weight (Cin, Cout, 4, 4) fp32 -> bf16 [Cout][tap*Cin+ci] (gather form)
__global__ void k_wconvt(const float* src, u16b* dst, int Cin, int Cout) {
    int i = blockIdx.x * blockDim.x + threadIdx.x;
    int K = 16 * Cin;
    if (i >= Cout * K) return;
    int co = i / K; int k = i - co * K;
    int tap = k / Cin; int ci = k - tap * Cin;
    int ky = tap >> 2; int kx = tap & 3;
    dst[i] = f2bf(src[(((size_t)ci * Cout + co) * 4 + ky) * 4 + kx]);
}

// ---------------------------------------------------------------------------
// Vector quantization: argmin_k ||e_k||^2 - 2 z.e_k over K=512 codes, C=3
// ---------------------------------------------------------------------------
__global__ void k_vq(const u16b* ze, const float* emb, u16b* zq_nhwc, float* zq_nchw,
                     int N, int H, int W, int K) {
    __shared__ float4 se[512];
    for (int k = threadIdx.x; k < K; k += blockDim.x) {
        float e0 = emb[k * 3 + 0], e1 = emb[k * 3 + 1], e2 = emb[k * 3 + 2];
        se[k] = make_float4(e0, e1, e2, e0 * e0 + e1 * e1 + e2 * e2);
    }
    __syncthreads();
    int loc = blockIdx.x * blockDim.x + threadIdx.x;
    int L = N * H * W;
    if (loc >= L) return;
    float z0 = bf2f(ze[loc * 3 + 0]);
    float z1 = bf2f(ze[loc * 3 + 1]);
    float z2 = bf2f(ze[loc * 3 + 2]);
    int best = 0; float bd = 3.4e38f;
    for (int k = 0; k < K; ++k) {
        float4 e = se[k];
        float d = e.w - 2.f * (z0 * e.x + z1 * e.y + z2 * e.z);
        if (d < bd) { bd = d; best = k; }
    }
    float4 e = se[best];
    zq_nhwc[loc * 3 + 0] = f2bf(e.x);
    zq_nhwc[loc * 3 + 1] = f2bf(e.y);
    zq_nhwc[loc * 3 + 2] = f2bf(e.z);
    int hw = H * W;
    int n = loc / hw, r = loc - n * hw;
    float* o = zq_nchw + (size_t)n * 3 * hw + r;
    o[0] = e.x; o[hw] = e.y; o[2 * hw] = e.z;
}

// ---------------------------------------------------------------------------
// Host orchestration
// ---------------------------------------------------------------------------
extern "C" void kernel_launch(void* const* d_in, const int* in_sizes, int n_in,
                              void* d_out, int out_size, void* d_ws, size_t ws_size,
                              hipStream_t stream) {
    (void)in_sizes; (void)n_in; (void)out_size; (void)ws_size;
    const int N = 16, D = 512, ZC = 3, K = 512;

    char* base = (char*)d_ws;
    size_t off = 0;
    auto alloc = [&](size_t bytes) -> char* {
        off = (off + 255) & ~(size_t)255;
        char* p = base + off;
        off += bytes;
        return p;
    };

    // ---- activation buffers (bf16 NHWC) ----
    u16b* xin = (u16b*)alloc((size_t)N * 256 * 256 * 3 * 2);
    u16b* big = (u16b*)alloc((size_t)N * 128 * 128 * D * 2);
    u16b* s0  = (u16b*)alloc((size_t)N * 64 * 64 * D * 2);
    u16b* s1  = (u16b*)alloc((size_t)N * 64 * 64 * D * 2);
    u16b* s2  = (u16b*)alloc((size_t)N * 64 * 64 * D * 2);
    u16b* z0  = (u16b*)alloc((size_t)N * 64 * 64 * ZC * 2);
    u16b* z1  = (u16b*)alloc((size_t)N * 64 * 64 * ZC * 2);
    u16b* z2  = (u16b*)alloc((size_t)N * 256 * 256 * ZC * 2);

    // ---- weight conversion ----
    auto convW = [&](int idx, int Cout, int Cin, int KH, int KW) -> u16b* {
        int Kt = KH * KW * Cin;
        u16b* dst = (u16b*)alloc((size_t)Cout * Kt * 2);
        int tot = Cout * Kt;
        k_wconv<<<(tot + 255) / 256, 256, 0, stream>>>((const float*)d_in[idx], dst, Cout, Cin, KH, KW);
        return dst;
    };
    auto convWT = [&](int idx, int Cin, int Cout) -> u16b* {
        int Kt = 16 * Cin;
        u16b* dst = (u16b*)alloc((size_t)Cout * Kt * 2);
        int tot = Cout * Kt;
        k_wconvt<<<(tot + 255) / 256, 256, 0, stream>>>((const float*)d_in[idx], dst, Cin, Cout);
        return dst;
    };
    auto B = [&](int idx) -> const float* { return (const float*)d_in[idx]; };

    u16b* w_enc0  = convW(1, D, 3, 4, 4);
    u16b* w_enc1  = convW(3, D, D, 4, 4);
    u16b* w_enc2  = convW(5, D, D, 3, 3);
    u16b* w_r1c1  = convW(7, D, D, 3, 3);
    u16b* w_r1c2  = convW(9, D, D, 1, 1);
    u16b* w_r2c1  = convW(11, D, D, 3, 3);
    u16b* w_r2c2  = convW(13, D, D, 1, 1);
    u16b* w_eout  = convW(15, ZC, D, 3, 3);
    u16b* w_preq  = convW(17, ZC, ZC, 1, 1);
    u16b* w_postq = convW(20, ZC, ZC, 1, 1);
    u16b* w_decin = convW(22, D, ZC, 3, 3);
    u16b* w_dec0  = convW(24, D, D, 3, 3);
    u16b* w_d1c1  = convW(26, D, D, 3, 3);
    u16b* w_d1c2  = convW(28, D, D, 1, 1);
    u16b* w_d2c1  = convW(30, D, D, 3, 3);
    u16b* w_d2c2  = convW(32, D, D, 1, 1);
    u16b* w_dt1   = convWT(34, D, D);
    u16b* w_dt2   = convWT(36, D, ZC);

    auto conv = [&](const u16b* A, const u16b* Wt, const float* bias, const u16b* R, u16b* Dst,
                    int IH, int IW, int Cin, int OH, int OW, int Cout,
                    int KH, int KW, int st, int pad, int ri, int ro, bool trans) {
        GP p;
        p.A = A; p.W = Wt; p.bias = bias; p.R = R; p.D = Dst;
        p.N = N; p.IH = IH; p.IW = IW; p.Cin = Cin; p.OH = OH; p.OW = OW; p.Cout = Cout;
        p.KH = KH; p.KW = KW; p.stride = st; p.pad = pad;
        p.Ktot = KH * KW * Cin; p.Mtot = N * OH * OW;
        p.relu_in = ri; p.relu_out = ro;
        dim3 g((p.Mtot + 127) / 128, (Cout + 127) / 128), b(256);
        if (trans) k_conv_gemm<1><<<g, b, 0, stream>>>(p);
        else       k_conv_gemm<0><<<g, b, 0, stream>>>(p);
    };

    // ---- input to NHWC bf16 ----
    {
        int tot = N * 256 * 256 * 3;
        k_tonhwc<<<(tot + 255) / 256, 256, 0, stream>>>((const float*)d_in[0], xin, N, 3, 256, 256);
    }

    // ---- encoder ----
    conv(xin, w_enc0, B(2), nullptr, big, 256, 256, 3, 128, 128, D, 4, 4, 2, 1, 0, 1, false);
    conv(big, w_enc1, B(4), nullptr, s0, 128, 128, D, 64, 64, D, 4, 4, 2, 1, 0, 1, false);
    conv(s0, w_enc2, B(6), nullptr, s1, 64, 64, D, 64, 64, D, 3, 3, 1, 1, 0, 0, false);
    // enc residual 1
    conv(s1, w_r1c1, B(8), nullptr, s2, 64, 64, D, 64, 64, D, 3, 3, 1, 1, 1, 1, false);
    conv(s2, w_r1c2, B(10), s1, s0, 64, 64, D, 64, 64, D, 1, 1, 1, 0, 0, 0, false);
    // enc residual 2
    conv(s0, w_r2c1, B(12), nullptr, s2, 64, 64, D, 64, 64, D, 3, 3, 1, 1, 1, 1, false);
    conv(s2, w_r2c2, B(14), s0, s1, 64, 64, D, 64, 64, D, 1, 1, 1, 0, 0, 0, false);
    // enc_out + pre_q
    conv(s1, w_eout, B(16), nullptr, z0, 64, 64, D, 64, 64, ZC, 3, 3, 1, 1, 0, 0, false);
    conv(z0, w_preq, B(18), nullptr, z1, 64, 64, ZC, 64, 64, ZC, 1, 1, 1, 0, 0, 0, false);

    // ---- outputs: ze (fp32 NCHW), vq -> zq ----
    float* out_xhat = (float*)d_out;
    float* out_ze   = out_xhat + (size_t)N * 3 * 256 * 256;
    float* out_zq   = out_ze + (size_t)N * ZC * 64 * 64;
    {
        int tot = N * ZC * 64 * 64;
        k_tonchw<<<(tot + 255) / 256, 256, 0, stream>>>(z1, out_ze, N, ZC, 64, 64);
        int L = N * 64 * 64;
        k_vq<<<(L + 255) / 256, 256, 0, stream>>>(z1, (const float*)d_in[19], z0, out_zq, N, 64, 64, K);
    }

    // ---- decoder (straight-through: consumes zq) ----
    conv(z0, w_postq, B(21), nullptr, z1, 64, 64, ZC, 64, 64, ZC, 1, 1, 1, 0, 0, 0, false);
    conv(z1, w_decin, B(23), nullptr, s0, 64, 64, ZC, 64, 64, D, 3, 3, 1, 1, 0, 0, false);
    conv(s0, w_dec0, B(25), nullptr, s1, 64, 64, D, 64, 64, D, 3, 3, 1, 1, 0, 0, false);
    // dec residual 1
    conv(s1, w_d1c1, B(27), nullptr, s2, 64, 64, D, 64, 64, D, 3, 3, 1, 1, 1, 1, false);
    conv(s2, w_d1c2, B(29), s1, s0, 64, 64, D, 64, 64, D, 1, 1, 1, 0, 0, 0, false);
    // dec residual 2
    conv(s0, w_d2c1, B(31), nullptr, s2, 64, 64, D, 64, 64, D, 3, 3, 1, 1, 1, 1, false);
    conv(s2, w_d2c2, B(33), s0, s1, 64, 64, D, 64, 64, D, 1, 1, 1, 0, 0, 0, false);
    // transposed convs
    conv(s1, w_dt1, B(35), nullptr, big, 64, 64, D, 128, 128, D, 4, 4, 2, 1, 0, 1, true);
    conv(big, w_dt2, B(37), nullptr, z2, 128, 128, D, 256, 256, ZC, 4, 4, 2, 1, 0, 0, true);

    // ---- x_hat to fp32 NCHW ----
    {
        int tot = N * 3 * 256 * 256;
        k_tonchw<<<(tot + 255) / 256, 256, 0, stream>>>(z2, out_xhat, N, 3, 256, 256);
    }
}
